// BaseLayer_55336358643333
// MI455X (gfx1250) — matrix operand order repaired and checked by reading, not compiled, not run on hardware
//
#include <hip/hip_runtime.h>
#include <hip/hip_bf16.h>

// MoE BaseLayer for MI455X (gfx1250): top-1 routing + grouped 32-row-tile FFN.
// GEMMs run on the bf16 WMMA unit (V_WMMA_F32_16X16X32_BF16) with a
// split-precision (bf16-hi + bf16-lo, 3-product) scheme for ~fp32 accuracy.
// Each wave owns 2 M-tiles so every converted weight fragment feeds 6 WMMAs,
// halving the fp32->bf16 conversion VALU per unit of matrix math.
// E=8, D=512, F=2048, T=2048.

#define D_DIM 512
#define F_DIM 2048
#define E_NUM 8
#define T_TOK 2048

// LDS row stride: 260 dwords (= 520 bf16 = 512 + 8 pad).
//  - multiple of 4 dwords -> 16B-aligned ds_load_b128
//  - 260 mod 64 == 4 -> half-wave b128 A-fragment reads tile all 64 banks
#define SWD 260

typedef __attribute__((ext_vector_type(8)))  float        v8f;
typedef __attribute__((ext_vector_type(16))) __bf16       v16bf;
typedef __attribute__((ext_vector_type(4)))  unsigned int v4u;

struct Q2 { v4u a, b; };   // 32B: two b128 LDS loads -> one v16bf A-fragment

__device__ __forceinline__ v8f wmma_bf(v16bf a, v16bf b, v8f c) {
  // D(16x16 f32) = A(16x32 bf16) * B(32x16 bf16) + C
  return __builtin_amdgcn_wmma_f32_16x16x32_bf16(
      /*neg_a=*/false, a, /*neg_b=*/false, b,
      /*c_mod=*/(short)0, c, /*reuse_a=*/false, /*reuse_b=*/false);
}

__device__ __forceinline__ unsigned int pack2(__bf16 lo_half, __bf16 hi_half) {
  const unsigned short a = __builtin_bit_cast(unsigned short, lo_half);
  const unsigned short b = __builtin_bit_cast(unsigned short, hi_half);
  return ((unsigned int)b << 16) | a;   // [15:0]=even K, [31:16]=odd K
}

// Load one 16-bf16 A-fragment (two contiguous 16B runs) from an LDS plane.
__device__ __forceinline__ v16bf ld_afrag(const unsigned int* buf, int dw) {
  Q2 q;
  q.a = *(const v4u*)(buf + dw);
  q.b = *(const v4u*)(buf + dw + 4);
  return __builtin_bit_cast(v16bf, q);
}

// Build bf16 hi/lo B-fragments from 16 strided fp32 weight rows.
__device__ __forceinline__ void ld_bfrag(const float* __restrict__ wp, size_t rowstride,
                                         v16bf& bh, v16bf& bl) {
  float w[16];
#pragma unroll
  for (int j = 0; j < 16; ++j) w[j] = wp[(size_t)j * rowstride];
#pragma unroll
  for (int j = 0; j < 16; ++j) {
    const __bf16 h = (__bf16)w[j];
    bh[j] = h;
    bl[j] = (__bf16)(w[j] - (float)h);
  }
}

// ---------------------------------------------------------------- init
__global__ void moe_init(int* __restrict__ counts, int* __restrict__ fill) {
  const int i = threadIdx.x;
  if (i < E_NUM) { counts[i] = 0; fill[i] = 0; }
}

// ---------------------------------------------------------------- router
__global__ void moe_router(const float* __restrict__ x,
                           const float* __restrict__ cent,
                           int* __restrict__ counts,
                           int* __restrict__ assign,
                           float* __restrict__ alpha) {
  const int t = blockIdx.x * blockDim.x + threadIdx.x;
  if (t >= T_TOK) return;
  const float4* xr = (const float4*)(x + (size_t)t * D_DIM);
  float s[E_NUM];
#pragma unroll
  for (int e = 0; e < E_NUM; ++e) s[e] = 0.f;
  for (int d4 = 0; d4 < D_DIM / 4; ++d4) {
    const float4 xv = xr[d4];
#pragma unroll
    for (int e = 0; e < E_NUM; ++e) {
      const float4 cv = ((const float4*)(cent + e * D_DIM))[d4];
      s[e] = fmaf(xv.x, cv.x, fmaf(xv.y, cv.y, fmaf(xv.z, cv.z, fmaf(xv.w, cv.w, s[e]))));
    }
  }
  int best = 0; float bs = s[0];
#pragma unroll
  for (int e = 1; e < E_NUM; ++e) { if (s[e] > bs) { bs = s[e]; best = e; } }
  assign[t] = best;
  alpha[t]  = 1.0f / (1.0f + expf(-bs));
  atomicAdd(&counts[best], 1);
}

// ---------------------------------------------------------------- offsets
__global__ void moe_offsets(const int* __restrict__ counts, int* __restrict__ offsets) {
  if (threadIdx.x == 0) {
    int acc = 0;
    for (int e = 0; e < E_NUM; ++e) { offsets[e] = acc; acc += counts[e]; }
    offsets[E_NUM] = acc;
  }
}

// ---------------------------------------------------------------- scatter
__global__ void moe_scatter(const int* __restrict__ assign,
                            const int* __restrict__ offsets,
                            int* __restrict__ fill,
                            int* __restrict__ sorted) {
  const int t = blockIdx.x * blockDim.x + threadIdx.x;
  if (t >= T_TOK) return;
  const int e = assign[t];
  const int pos = offsets[e] + atomicAdd(&fill[e], 1);
  sorted[pos] = t;
}

// ---------------------------------------------------------------- grouped FFN
// grid.x = E_NUM*64, one 32-token tile per block, 8 wave32's.
// Each wave: 4 N-tiles x 2 M-tiles. LDS planes (dwords): hhi/hlo = LN(x)
// split across 32 rows, ahi/alo = relu(ff1) chunk split across 32 rows.
__global__ __launch_bounds__(256)
void moe_expert_ffn(const float* __restrict__ x,
                    const float* __restrict__ ln_g, const float* __restrict__ ln_b,
                    const float* __restrict__ W1,   const float* __restrict__ b1,
                    const float* __restrict__ W2,   const float* __restrict__ b2,
                    const int* __restrict__ counts, const int* __restrict__ offsets,
                    const int* __restrict__ sorted, const float* __restrict__ alpha,
                    float* __restrict__ out) {
  const int e = blockIdx.x >> 6;
  const int tileIdx = blockIdx.x & 63;
  const int cnt = counts[e];
  if (tileIdx * 32 >= cnt) return;       // uniform early-exit (EXEC stays full)
  const int base = offsets[e] + tileIdx * 32;

  extern __shared__ unsigned int smem[];
  unsigned int* hhi = smem;              // 32 x SWD dwords
  unsigned int* hlo = smem + 32 * SWD;
  unsigned int* ahi = smem + 64 * SWD;
  unsigned int* alo = smem + 96 * SWD;
  __shared__ int   rowTok[32];
  __shared__ float rowAlpha[32];

  const int lane  = threadIdx.x & 31;
  const int wv    = threadIdx.x >> 5;
  const int m16   = lane & 15;          // A-row / B-col / C-col
  const int g     = lane >> 4;          // half-wave: K sub-block select
  const int mbase = g << 3;             // C rows 0..7 or 8..15

  // ---- Phase A: gather + LayerNorm, split into bf16 hi/lo planes ----
  for (int rr = 0; rr < 4; ++rr) {
    const int r = wv * 4 + rr;
    const bool valid = (tileIdx * 32 + r) < cnt;
    const int tok = valid ? sorted[base + r] : -1;
    float v0[8], v1[8];
    float sum = 0.f, sq = 0.f;
    if (valid) {
      const float* xr = x + (size_t)tok * D_DIM;
#pragma unroll
      for (int i = 0; i < 8; ++i) {
        const int c = 2 * lane + 64 * i;
        const float a = xr[c], b = xr[c + 1];
        v0[i] = a; v1[i] = b;
        sum += a + b;
        sq = fmaf(a, a, fmaf(b, b, sq));
      }
    } else {
#pragma unroll
      for (int i = 0; i < 8; ++i) { v0[i] = 0.f; v1[i] = 0.f; }
    }
#pragma unroll
    for (int off = 16; off > 0; off >>= 1) {
      sum += __shfl_xor(sum, off, 32);
      sq  += __shfl_xor(sq,  off, 32);
    }
    const float mu   = sum * (1.0f / D_DIM);
    const float var  = sq * (1.0f / D_DIM) - mu * mu;
    const float rstd = rsqrtf(var + 1e-5f);
#pragma unroll
    for (int i = 0; i < 8; ++i) {
      const int c = 2 * lane + 64 * i;
      const float h0 = valid
        ? fmaf((v0[i] - mu) * rstd, ln_g[e * D_DIM + c],     ln_b[e * D_DIM + c])     : 0.f;
      const float h1 = valid
        ? fmaf((v1[i] - mu) * rstd, ln_g[e * D_DIM + c + 1], ln_b[e * D_DIM + c + 1]) : 0.f;
      const __bf16 h0h = (__bf16)h0, h1h = (__bf16)h1;
      const __bf16 h0l = (__bf16)(h0 - (float)h0h);
      const __bf16 h1l = (__bf16)(h1 - (float)h1h);
      const int idx = r * SWD + lane + 32 * i;   // dword = K-pair index
      hhi[idx] = pack2(h0h, h1h);
      hlo[idx] = pack2(h0l, h1l);
    }
    if (lane == 0) { rowTok[r] = tok; rowAlpha[r] = valid ? alpha[tok] : 0.f; }
  }
  __syncthreads();

  v8f oacc[4][2];                        // [N-tile][M-tile], persists over chunks
#pragma unroll
  for (int t = 0; t < 4; ++t) { v8f z = {}; oacc[t][0] = z; oacc[t][1] = z; }

  const size_t w1base = (size_t)e * D_DIM * F_DIM;
  const size_t w2base = (size_t)e * F_DIM * D_DIM;
  __bf16* aHi = (__bf16*)ahi;
  __bf16* aLo = (__bf16*)alo;

  // ---- F in 4 chunks of 512: FF1 -> bf16-split LDS -> FF2 partial ----
  for (int fc = 0; fc < 4; ++fc) {
    // FF1 in two N-halves (2 N-tiles each) to bound register pressure
    for (int half = 0; half < 2; ++half) {
      v8f acc[2][2];                     // [N-tile][M-tile]
#pragma unroll
      for (int t = 0; t < 2; ++t) { v8f z = {}; acc[t][0] = z; acc[t][1] = z; }
      for (int kb = 0; kb < D_DIM; kb += 32) {
        const int adw0 = m16 * SWD + (kb >> 1) + g * 8;
        const int adw1 = adw0 + 16 * SWD;
        const v16bf ah0 = ld_afrag(hhi, adw0);
        const v16bf al0 = ld_afrag(hlo, adw0);
        const v16bf ah1 = ld_afrag(hhi, adw1);
        const v16bf al1 = ld_afrag(hlo, adw1);
#pragma unroll
        for (int t = 0; t < 2; ++t) {
          const int n = fc * 512 + wv * 64 + (half * 2 + t) * 16 + m16;
          const float* wp = W1 + w1base + (size_t)(kb + g * 16) * F_DIM + n;
          v16bf bh, bl;
          ld_bfrag(wp, F_DIM, bh, bl);   // converted once, feeds 6 WMMAs
          acc[t][0] = wmma_bf(ah0, bh, acc[t][0]);
          acc[t][0] = wmma_bf(al0, bh, acc[t][0]);
          acc[t][0] = wmma_bf(ah0, bl, acc[t][0]);
          acc[t][1] = wmma_bf(ah1, bh, acc[t][1]);
          acc[t][1] = wmma_bf(al1, bh, acc[t][1]);
          acc[t][1] = wmma_bf(ah1, bl, acc[t][1]);
        }
      }
      // bias + relu + split-store into act planes (both M-tiles)
#pragma unroll
      for (int t = 0; t < 2; ++t) {
        const int ncol = wv * 64 + (half * 2 + t) * 16 + m16;
        const float bias = b1[e * F_DIM + fc * 512 + ncol];
#pragma unroll
        for (int mt = 0; mt < 2; ++mt) {
#pragma unroll
          for (int v = 0; v < 8; ++v) {
            float val = acc[t][mt][v] + bias;
            val = val > 0.f ? val : 0.f;
            const __bf16 vh = (__bf16)val;
            const int idx = (mt * 16 + mbase + v) * (2 * SWD) + ncol; // bf16 units
            aHi[idx] = vh;
            aLo[idx] = (__bf16)(val - (float)vh);
          }
        }
      }
    }
    __syncthreads();

    // FF2 partial: oacc += act_chunk @ W2[chunk, :]
    for (int kb = 0; kb < 512; kb += 32) {
      const int adw0 = m16 * SWD + (kb >> 1) + g * 8;
      const int adw1 = adw0 + 16 * SWD;
      const v16bf ah0 = ld_afrag(ahi, adw0);
      const v16bf al0 = ld_afrag(alo, adw0);
      const v16bf ah1 = ld_afrag(ahi, adw1);
      const v16bf al1 = ld_afrag(alo, adw1);
      const size_t kg = (size_t)(fc * 512 + kb + g * 16);
#pragma unroll
      for (int t = 0; t < 4; ++t) {
        const int n = wv * 64 + t * 16 + m16;
        const float* wp = W2 + w2base + kg * D_DIM + n;
        v16bf bh, bl;
        ld_bfrag(wp, D_DIM, bh, bl);
        oacc[t][0] = wmma_bf(ah0, bh, oacc[t][0]);
        oacc[t][0] = wmma_bf(al0, bh, oacc[t][0]);
        oacc[t][0] = wmma_bf(ah0, bl, oacc[t][0]);
        oacc[t][1] = wmma_bf(ah1, bh, oacc[t][1]);
        oacc[t][1] = wmma_bf(al1, bh, oacc[t][1]);
        oacc[t][1] = wmma_bf(ah1, bl, oacc[t][1]);
      }
    }
    __syncthreads();
  }

  // ---- Epilogue: out[t] = x[t] + alpha * (ffn + b2) ----
#pragma unroll
  for (int t = 0; t < 4; ++t) {
    const int n = wv * 64 + t * 16 + m16;
    const float bias = b2[e * D_DIM + n];
#pragma unroll
    for (int mt = 0; mt < 2; ++mt) {
#pragma unroll
      for (int v = 0; v < 8; ++v) {
        const int tok = rowTok[mt * 16 + mbase + v];
        if (tok >= 0) {
          const float al = rowAlpha[mt * 16 + mbase + v];
          const float xv = x[(size_t)tok * D_DIM + n];
          out[(size_t)tok * D_DIM + n] = fmaf(al, oacc[t][mt][v] + bias, xv);
        }
      }
    }
  }
}

// ---------------------------------------------------------------- launch
extern "C" void kernel_launch(void* const* d_in, const int* in_sizes, int n_in,
                              void* d_out, int out_size, void* d_ws, size_t ws_size,
                              hipStream_t stream) {
  const float* x    = (const float*)d_in[0];
  const float* cent = (const float*)d_in[1];
  const float* ln_g = (const float*)d_in[2];
  const float* ln_b = (const float*)d_in[3];
  const float* W1   = (const float*)d_in[4];
  const float* b1   = (const float*)d_in[5];
  const float* W2   = (const float*)d_in[6];
  const float* b2   = (const float*)d_in[7];
  float* out = (float*)d_out;

  int*   counts  = (int*)d_ws;
  int*   fill    = counts + 8;
  int*   offsets = fill + 8;
  int*   assign  = offsets + 16;
  int*   sorted  = assign + T_TOK;
  float* alphas  = (float*)(sorted + T_TOK);

  moe_init   <<<1, 32, 0, stream>>>(counts, fill);
  moe_router <<<T_TOK / 256, 256, 0, stream>>>(x, cent, counts, assign, alphas);
  moe_offsets<<<1, 32, 0, stream>>>(counts, offsets);
  moe_scatter<<<T_TOK / 256, 256, 0, stream>>>(assign, offsets, fill, sorted);

  const size_t smem = (size_t)(128 * SWD) * sizeof(unsigned int); // 133,120 B < 320 KB/WGP
  moe_expert_ffn<<<dim3(E_NUM * 64), 256, smem, stream>>>(
      x, ln_g, ln_b, W1, b1, W2, b2, counts, offsets, sorted, alphas, out);
}
